// MultiHeadCulturalAttention_68685116998268
// MI455X (gfx1250) — matrix operand order repaired and checked by reading, not compiled, not run on hardware
//
#include <hip/hip_runtime.h>
#include <hip/hip_bf16.h>
#include <math.h>

typedef __bf16 bf16;
typedef __bf16 v16bf __attribute__((ext_vector_type(16)));
typedef __bf16 v8bf  __attribute__((ext_vector_type(8)));
typedef float  v8f   __attribute__((ext_vector_type(8)));
typedef int    v4i   __attribute__((ext_vector_type(4)));

#define E_DIM 768
#define AS1 __attribute__((address_space(1)))
#define AS3 __attribute__((address_space(3)))

// CDNA5 async global->LDS path (ASYNCcnt), guarded so the file still builds
// if this toolchain doesn't expose the builtins.
#if defined(__has_builtin)
#if __has_builtin(__builtin_amdgcn_global_load_async_to_lds_b128) && \
    __has_builtin(__builtin_amdgcn_s_wait_asynccnt)
#define USE_ASYNC 1
#endif
#endif
#ifndef USE_ASYNC
#define USE_ASYNC 0
#endif

// ---------------------------------------------------------------------------
// fp32 -> bf16 elementwise convert
// ---------------------------------------------------------------------------
__global__ void cvt_f32_bf16_kernel(const float* __restrict__ s,
                                    bf16* __restrict__ d, int n) {
  int i = blockIdx.x * 256 + threadIdx.x;
  if (i < n) d[i] = (bf16)s[i];
}

// ---------------------------------------------------------------------------
// Tiled WMMA GEMM: C[M,N] = A[M,K](bf16) @ W[K,N](bf16, row-major) + bias(+bias2)
// Block = 256 threads = 8 waves; tile 128(M) x 64(N); K-step 32.
// B tile double-buffered in LDS; staged with async global->LDS when available
// (prefetch tile k+1 while WMMAs consume tile k, then s_wait_asynccnt 1).
// Fragment layouts (CDNA5 ISA):
//   A 16x32 bf16 : lane -> M = lane%16 ; element e -> K = e + (e>=8?8:0) + (lane>=16?8:0)
//   B 32x16 bf16 : lane -> N = lane%16 ; element e -> K = same as A (strided LDS reads)
//   C 16x16 f32  : element r -> M = r + 8*(lane>=16) ; N = lane%16
// ---------------------------------------------------------------------------
template<bool OUT_F32>
__global__ void __launch_bounds__(256)
gemm_bias_kernel(const bf16* __restrict__ A, int lda,
                 const bf16* __restrict__ Bw, int ldb,
                 const float* __restrict__ bias,
                 const float* __restrict__ bias2,
                 void* Cp, int ldc, int ccol0,
                 int N, int K)
{
  __shared__ bf16 sB[2][32 * 64];          // 2 x 4 KB B tiles
  const int tid  = threadIdx.x;
  const int wave = tid >> 5;
  const int lane = tid & 31;
  const int l16  = lane & 15;
  const int lh   = lane >> 4;
  const int nTiles = N / 64;
  const int m0 = (blockIdx.x / nTiles) * 128 + wave * 16;
  const int n0 = (blockIdx.x % nTiles) * 64;

  // each thread owns one 16B chunk of the 32x64 B tile
  const int br = tid >> 3;
  const int bc = (tid & 7) * 8;

  auto stage = [&](int ks, int buf) {
    const bf16* src = &Bw[(size_t)(ks * 32 + br) * ldb + n0 + bc];
    bf16* dst = &sB[buf][br * 64 + bc];
#if USE_ASYNC
    __builtin_amdgcn_global_load_async_to_lds_b128(
        (AS1 v4i*)src, (AS3 v4i*)dst, 0, 0);
#else
    *(v8bf*)dst = *(const v8bf*)src;
#endif
  };

  v8f acc[4] = {};
  const int nk = K / 32;
  stage(0, 0);
  for (int ks = 0; ks < nk; ++ks) {
    const int buf = ks & 1;
    const bool pre = (ks + 1) < nk;
    if (pre) stage(ks + 1, buf ^ 1);       // overlap next tile with this compute
#if USE_ASYNC
    if (pre) __builtin_amdgcn_s_wait_asynccnt(1);
    else     __builtin_amdgcn_s_wait_asynccnt(0);
#endif
    __syncthreads();                        // all waves' chunks of buf present

    v16bf af;
    const bf16* arow = A + (size_t)(m0 + l16) * lda + ks * 32 + lh * 8;
    *(v8bf*)&af       = *(const v8bf*)arow;         // K = lh*8 + 0..7
    *((v8bf*)&af + 1) = *(const v8bf*)(arow + 16);  // K = lh*8 + 16..23

#pragma unroll
    for (int nf = 0; nf < 4; ++nf) {
      v16bf bfg;
      for (int e = 0; e < 16; ++e)
        bfg[e] = sB[buf][(lh * 8 + e + (e >= 8 ? 8 : 0)) * 64 + nf * 16 + l16];
      acc[nf] = __builtin_amdgcn_wmma_f32_16x16x32_bf16(
          false, af, false, bfg, (short)0, acc[nf], false, false);
    }
    __syncthreads();                        // reads done before buf is restaged
  }

#pragma unroll
  for (int nf = 0; nf < 4; ++nf) {
#pragma unroll
    for (int r = 0; r < 8; ++r) {
      int m = m0 + r + 8 * lh;
      int n = n0 + nf * 16 + l16;
      float v = acc[nf][r];
      if (bias)  v += bias[n];
      if (bias2) v += bias2[n];
      if (OUT_F32) ((float*)Cp)[(size_t)m * ldc + ccol0 + n] = v;
      else         ((bf16*)Cp)[(size_t)m * ldc + ccol0 + n] = (bf16)v;
    }
  }
}

// ---------------------------------------------------------------------------
// Flash attention (online softmax), one wave per 16 query rows.
// Launch guarantees all 8 waves of a block share (b,h): S/16 tiles is a
// multiple of 8, so a block covers 8 consecutive t-tiles of one head.
// That lets the block cooperatively stage K^T for each 32-column s-block in
// LDS (contiguous 16B global loads + scalar stores in [s][d] order, amortized 8x);
// score B-frags then become two ds_load_b128 per fragment.
// P round-trips through a per-wave LDS slice (C-layout -> A-layout).
// V B-frags load from global, strided by E_DIM (lane%16 = d col, elements = s rows).
// Each launch covers output channels [d0, d0+128) (8 accumulator frags).
// ---------------------------------------------------------------------------
template<int NH, int HD, bool CULTURAL>
__global__ void __launch_bounds__(256)
attn_kernel(const bf16* __restrict__ Q, const bf16* __restrict__ Km,
            const bf16* __restrict__ V,
            const float* __restrict__ amask,   // [B, S]
            const float* __restrict__ cmask,   // [B, S, S] or nullptr
            bf16* __restrict__ ctx,            // [B*S, 768]
            int d0, float scale, int S)
{
  __shared__ bf16 sP[8][16 * 32];            // per-wave P tile
  __shared__ bf16 sKT[HD * 32];              // K tile: [s_local][d], shared
  const int tid  = threadIdx.x;
  const int lane = tid & 31;
  const int wave = tid >> 5;
  const int l16  = lane & 15;
  const int lh   = lane >> 4;
  const int wv   = blockIdx.x * 8 + wave;
  const int nT   = S / 16;
  const int t0   = (wv % nT) * 16;
  const int h    = (wv / nT) % NH;
  const int b    = wv / (nT * NH);

  constexpr int NKT = HD / 32;
  v16bf qf[NKT];
  {
    const bf16* qrow = Q + (size_t)(b * S + t0 + l16) * E_DIM + h * HD + lh * 8;
#pragma unroll
    for (int kt = 0; kt < NKT; ++kt) {
      *(v8bf*)&qf[kt]       = *(const v8bf*)(qrow + kt * 32);
      *((v8bf*)&qf[kt] + 1) = *(const v8bf*)(qrow + kt * 32 + 16);
    }
  }

  v8f acc[8] = {};
  float mrun[8], lrun[8];
#pragma unroll
  for (int r = 0; r < 8; ++r) { mrun[r] = -1e30f; lrun[r] = 0.f; }

  for (int s0 = 0; s0 < S; s0 += 32) {
    // ---- cooperative K^T staging (WAR-safe: prior iteration's reads finished
    // before its end barrier, since the WMMAs forced s_wait_dscnt) -----------
    {
      const int sl = tid >> 3;               // 0..31 : s row
      const bf16* krow = Km + (size_t)(b * S + s0 + sl) * E_DIM + h * HD;
#pragma unroll
      for (int dc = (tid & 7) * 16; dc < HD; dc += 128) {
        v8bf a0 = *(const v8bf*)(krow + dc);
        v8bf a1 = *(const v8bf*)(krow + dc + 8);
#pragma unroll
        for (int j = 0; j < 8; ++j) sKT[sl * HD + dc + j] = a0[j];
#pragma unroll
        for (int j = 0; j < 8; ++j) sKT[sl * HD + dc + 8 + j] = a1[j];
      }
    }
    __syncthreads();

    // ---- scores: two 16x16 fragments (s columns s0..s0+31) ----------------
    float p[2][8];
#pragma unroll
    for (int ht = 0; ht < 2; ++ht) {
      v8f sc = {};
#pragma unroll
      for (int kt = 0; kt < NKT; ++kt) {
        v16bf kf;                             // lane%16 = s col, elements = d (table order)
        const bf16* kp = &sKT[(ht * 16 + l16) * HD + kt * 32 + lh * 8];
        *(v8bf*)&kf       = *(const v8bf*)kp;
        *((v8bf*)&kf + 1) = *(const v8bf*)(kp + 16);
        sc = __builtin_amdgcn_wmma_f32_16x16x32_bf16(
            false, qf[kt], false, kf, (short)0, sc, false, false);
      }
      float am = amask[(size_t)b * S + s0 + ht * 16 + l16];
#pragma unroll
      for (int r = 0; r < 8; ++r) {
        float v = sc[r] * scale + am;
        if (CULTURAL)
          v += cmask[((size_t)(b * S) + t0 + r + 8 * lh) * S + s0 + ht * 16 + l16];
        p[ht][r] = v;
      }
    }

    // ---- online softmax update (rows live on 16-lane groups) --------------
#pragma unroll
    for (int r = 0; r < 8; ++r) {
      float rm = fmaxf(p[0][r], p[1][r]);
#pragma unroll
      for (int off = 1; off < 16; off <<= 1) rm = fmaxf(rm, __shfl_xor(rm, off, 32));
      float mnew = fmaxf(mrun[r], rm);
      float corr = __expf(mrun[r] - mnew);
      mrun[r] = mnew;
      float p0 = __expf(p[0][r] - mnew);
      float p1 = __expf(p[1][r] - mnew);
      p[0][r] = p0; p[1][r] = p1;
      float ps = p0 + p1;
#pragma unroll
      for (int off = 1; off < 16; off <<= 1) ps += __shfl_xor(ps, off, 32);
      lrun[r] = lrun[r] * corr + ps;
#pragma unroll
      for (int dt = 0; dt < 8; ++dt) acc[dt][r] = acc[dt][r] * corr;
    }

    // ---- C-layout P -> LDS -> A-layout fragment (DS in-order per wave) ----
#pragma unroll
    for (int ht = 0; ht < 2; ++ht)
#pragma unroll
      for (int r = 0; r < 8; ++r)
        sP[wave][(r + 8 * lh) * 32 + ht * 16 + l16] = (bf16)p[ht][r];
    v16bf pf;
    const bf16* pb = &sP[wave][l16 * 32 + lh * 8];
    *(v8bf*)&pf       = *(const v8bf*)pb;
    *((v8bf*)&pf + 1) = *(const v8bf*)(pb + 16);

    // ---- acc += P @ V (V B-frag: lane%16 = d col, elements = s rows, table order) --
    const bf16* vb = V + (size_t)(b * S + s0 + lh * 8) * E_DIM + h * HD + d0 + l16;
#pragma unroll
    for (int dt = 0; dt < 8; ++dt) {
      v16bf vf;
      for (int e = 0; e < 16; ++e)
        vf[e] = vb[(size_t)(e + (e >= 8 ? 8 : 0)) * E_DIM + dt * 16];
      acc[dt] = __builtin_amdgcn_wmma_f32_16x16x32_bf16(
          false, pf, false, vf, (short)0, acc[dt], false, false);
    }
    __syncthreads();                         // end-of-iter: guard sKT restage
  }

#pragma unroll
  for (int dt = 0; dt < 8; ++dt)
#pragma unroll
    for (int r = 0; r < 8; ++r) {
      float o = acc[dt][r] / lrun[r];
      ctx[(size_t)(b * S + t0 + r + 8 * lh) * E_DIM + h * HD + d0 + dt * 16 + l16] = (bf16)o;
    }
}

// ---------------------------------------------------------------------------
extern "C" void kernel_launch(void* const* d_in, const int* in_sizes, int n_in,
                              void* d_out, int out_size, void* d_ws, size_t ws_size,
                              hipStream_t stream) {
  (void)in_sizes; (void)n_in; (void)out_size; (void)ws_size;
  constexpr int Bn = 2, S = 2048, E = E_DIM, M = Bn * S;

  const float* hidden = (const float*)d_in[0];
  const float* cmask  = (const float*)d_in[1];
  const float* amask  = (const float*)d_in[2];
  const float* wf[8];  for (int i = 0; i < 8; ++i) wf[i] = (const float*)d_in[3 + i];
  const float* bv[8];  for (int i = 0; i < 8; ++i) bv[i] = (const float*)d_in[11 + i];
  const float* r_cb = (const float*)d_in[19];   // [E] flattened per-channel Q bias
  const float* c_cb = (const float*)d_in[20];
  const float* outw = (const float*)d_in[21];   // [2E, E]
  const float* outb = (const float*)d_in[22];

  // workspace allocator (bf16)
  bf16* wp = (bf16*)d_ws;
  auto take = [&](size_t n) { bf16* r = wp; wp += n; return r; };
  bf16* Xb = take((size_t)M * E);
  bf16* Wb[8]; for (int i = 0; i < 8; ++i) Wb[i] = take((size_t)E * E);
  bf16* OWb  = take((size_t)2 * E * E);
  bf16* Qr   = take((size_t)M * E);
  bf16* Kr   = take((size_t)M * E);
  bf16* Vr   = take((size_t)M * E);
  bf16* Qc   = take((size_t)M * E);
  bf16* Kc   = take((size_t)M * E);
  bf16* Vc   = take((size_t)M * E);
  bf16* CtxR = take((size_t)M * E);
  bf16* CtxC = take((size_t)M * E);
  bf16* Comb = take((size_t)M * 2 * E);

  auto cvt = [&](const float* s, bf16* d, int n) {
    cvt_f32_bf16_kernel<<<(n + 255) / 256, 256, 0, stream>>>(s, d, n);
  };
  cvt(hidden, Xb, M * E);
  for (int i = 0; i < 8; ++i) cvt(wf[i], Wb[i], E * E);
  cvt(outw, OWb, 2 * E * E);

  auto gemm_bf = [&](const bf16* A, int lda, const bf16* Bw, int ldb,
                     const float* bias, const float* bias2,
                     bf16* C, int ldc, int ccol0, int Mm, int N, int K) {
    dim3 grid((Mm / 128) * (N / 64));
    gemm_bias_kernel<false><<<grid, 256, 0, stream>>>(
        A, lda, Bw, ldb, bias, bias2, (void*)C, ldc, ccol0, N, K);
  };

  // QKV projections (Wb: 0=rq 1=rk 2=rv 3=ro 4=cq 5=ck 6=cv 7=co)
  gemm_bf(Xb, E, Wb[0], E, bv[0], r_cb,    Qr, E, 0, M, E, E);
  gemm_bf(Xb, E, Wb[1], E, bv[1], nullptr, Kr, E, 0, M, E, E);
  gemm_bf(Xb, E, Wb[2], E, bv[2], nullptr, Vr, E, 0, M, E, E);
  gemm_bf(Xb, E, Wb[4], E, bv[4], c_cb,    Qc, E, 0, M, E, E);
  gemm_bf(Xb, E, Wb[5], E, bv[5], nullptr, Kc, E, 0, M, E, E);
  gemm_bf(Xb, E, Wb[6], E, bv[6], nullptr, Vc, E, 0, M, E, E);

  // attention: regular nh=6 hd=128 (one d-pass), cultural nh=2 hd=384 (3 passes)
  {
    int wavesR = Bn * 6 * (S / 16);
    attn_kernel<6, 128, false><<<wavesR / 8, 256, 0, stream>>>(
        Qr, Kr, Vr, amask, nullptr, CtxR, 0, 0.08838834764831845f, S);
    int wavesC = Bn * 2 * (S / 16);
    for (int d0 = 0; d0 < 384; d0 += 128)
      attn_kernel<2, 384, true><<<wavesC / 8, 256, 0, stream>>>(
          Qc, Kc, Vc, amask, cmask, CtxC, d0, 0.05103103630798287f, S);
  }

  // O-projections -> concatenated buffer, then final 1536->768 projection (fp32 out)
  gemm_bf(CtxR, E, Wb[3], E, bv[3], nullptr, Comb, 2 * E, 0, M, E, E);
  gemm_bf(CtxC, E, Wb[7], E, bv[7], nullptr, Comb, 2 * E, E, M, E, E);
  {
    dim3 grid((M / 128) * (E / 64));
    gemm_bias_kernel<true><<<grid, 256, 0, stream>>>(
        Comb, 2 * E, OWb, E, outb, nullptr, d_out, E, 0, E, 2 * E);
  }
}
